// LSTM_17446157156703
// MI455X (gfx1250) — matrix operand order repaired and checked by reading, not compiled, not run on hardware
//
#include <hip/hip_runtime.h>
#include <hip/hip_bf16.h>

// ---------------------------------------------------------------------------
// LSTM scan, T=16384, D=512, gates=2048.
// Phase A: Zx = X@Wx + b  as bf16 WMMA GEMM (parallel, ~34 GFLOP),
//          software-pipelined loads (overlap VMEM with v_wmma).
// Phase B: 16-workgroup cooperative scan; each WG holds its 128KB slice of
//          bf16 Wh resident in LDS (preloaded with one TENSOR_LOAD_TO_LDS
//          TDM descriptor), computes its 128 gate columns per step with
//          v_wmma_f32_16x16x32_bf16, exchanges the 1KB h vector through L2
//          with a device-scope sense-reversing barrier.
// ---------------------------------------------------------------------------

#define T_STEPS 16384
#define DMODEL  512
#define GATES   2048
#define NWG     16

typedef __attribute__((ext_vector_type(16))) __bf16 v16bf;
typedef __attribute__((ext_vector_type(8)))  float  v8f;
typedef __attribute__((ext_vector_type(4)))  unsigned int v4u;
typedef __attribute__((ext_vector_type(8)))  int v8i;
typedef __attribute__((ext_vector_type(4)))  int v4i;

// workspace layout (bytes)
#define OFF_ZX   ((size_t)0)            // 16384*2048*4   = 128 MiB f32
#define OFF_XF   ((size_t)134217728)    // 16384*512*2    = 16 MiB  bf16 A-frags
#define OFF_WXF  ((size_t)150994944)    // 512*2048*2     = 2 MiB   bf16 B-frags
#define OFF_WHF  ((size_t)153092096)    // 512*2048*2     = 2 MiB   bf16 B-frags
#define OFF_HBUF ((size_t)155189248)    // 2*512*2        = 2 KiB   double-buffered h (bf16)
#define OFF_BAR  ((size_t)155191296)    // 2 * u32        barrier count / phase

__device__ __forceinline__ unsigned short f2bf(float f) {
  unsigned u = __builtin_bit_cast(unsigned, f);
  unsigned r = (u + 0x7FFFu + ((u >> 16) & 1u)) >> 16;   // RNE
  return (unsigned short)r;
}

__device__ __forceinline__ v8f wmma_bf16(v16bf a, v16bf b, v8f c) {
  return __builtin_amdgcn_wmma_f32_16x16x32_bf16(false, a, false, b,
                                                 (short)0, c, false, false);
}

// --- pack X [T,512] f32 -> A-fragment-linear bf16 --------------------------
// A 16x32 bf16 frag: lane l: m=l&15, hi=l>>4; elem e: K = 8*hi + (e&7) + 16*(e>>3)
__global__ __launch_bounds__(256) void lstm_pack_x(const float* __restrict__ x,
                                                   unsigned short* __restrict__ xf) {
  int idx  = blockIdx.x * 256 + threadIdx.x;   // 16384*512 total
  int e    = idx & 15;
  int l    = (idx >> 4) & 31;
  int tile = idx >> 9;                         // tile = mt*16 + kt
  int kt   = tile & 15, mt = tile >> 4;
  int row  = mt * 16 + (l & 15);
  int hi   = l >> 4;
  int K    = kt * 32 + 8 * hi + (e & 7) + 16 * (e >> 3);
  xf[idx] = f2bf(x[row * DMODEL + K]);
}

// --- pack W [512,2048] f32 -> B-fragment-linear bf16 -----------------------
// B 32x16 bf16 frag: lane l: n=l&15, hi=l>>4; elem e: K = 8*hi + (e&7) + 16*(e>>3)
__global__ __launch_bounds__(256) void lstm_pack_w(const float* __restrict__ w,
                                                   unsigned short* __restrict__ wf) {
  int idx  = blockIdx.x * 256 + threadIdx.x;   // 512*2048 total
  int e    = idx & 15;
  int l    = (idx >> 4) & 31;
  int tile = idx >> 9;                         // tile = nt*16 + kt
  int kt   = tile & 15, nt = tile >> 4;
  int n    = nt * 16 + (l & 15);
  int K    = kt * 32 + 8 * (l >> 4) + (e & 7) + 16 * (e >> 3);
  wf[idx] = f2bf(w[K * GATES + n]);
}

// --- init: h0 -> hbuf parity 0 (bf16), zero the inter-WG barrier -----------
__global__ void lstm_init(const float* __restrict__ h0,
                          unsigned short* __restrict__ hbuf,
                          unsigned* __restrict__ bar) {
  int i = blockIdx.x * blockDim.x + threadIdx.x;
  if (i < DMODEL) {
    hbuf[i] = f2bf(h0[i]);
    hbuf[DMODEL + i] = 0;
  }
  if (i == 0) { bar[0] = 0u; bar[1] = 0u; }
}

// --- Phase A: Zx = X @ Wx + b   (bf16 WMMA, f32 accumulate, pipelined) -----
__global__ __launch_bounds__(256) void lstm_gemm_xw(
    const unsigned short* __restrict__ xf, const unsigned short* __restrict__ wf,
    const float* __restrict__ bias, float* __restrict__ zx) {
  int mt   = blockIdx.x;                 // 1024 M-tiles of 16 rows
  int wave = threadIdx.x >> 5;
  int lane = threadIdx.x & 31;
  int nt0  = blockIdx.y * 32 + wave * 4; // 4 N-tiles per wave
  const unsigned short* pa = xf + (size_t)(mt * 16) * 512 + lane * 16;
  const unsigned short* pb = wf + (size_t)(nt0 * 16) * 512 + lane * 16;

  v8f acc[4] = {};
  // software pipeline: current fragments in regs, next iteration's loads in
  // flight while the matrix pipe consumes the current ones.
  v16bf a  = *(const v16bf*)pa;
  v16bf b0 = *(const v16bf*)(pb);
  v16bf b1 = *(const v16bf*)(pb + 1 * 8192);
  v16bf b2 = *(const v16bf*)(pb + 2 * 8192);
  v16bf b3 = *(const v16bf*)(pb + 3 * 8192);
#pragma unroll
  for (int kt = 0; kt < 16; ++kt) {
    v16bf an = a, bn0 = b0, bn1 = b1, bn2 = b2, bn3 = b3;
    if (kt < 15) {
      an  = *(const v16bf*)(pa + (kt + 1) * 512);
      bn0 = *(const v16bf*)(pb + (kt + 1) * 512);
      bn1 = *(const v16bf*)(pb + 1 * 8192 + (kt + 1) * 512);
      bn2 = *(const v16bf*)(pb + 2 * 8192 + (kt + 1) * 512);
      bn3 = *(const v16bf*)(pb + 3 * 8192 + (kt + 1) * 512);
    }
    acc[0] = wmma_bf16(a, b0, acc[0]);
    acc[1] = wmma_bf16(a, b1, acc[1]);
    acc[2] = wmma_bf16(a, b2, acc[2]);
    acc[3] = wmma_bf16(a, b3, acc[3]);
    a = an; b0 = bn0; b1 = bn1; b2 = bn2; b3 = bn3;
  }
  int hi = lane >> 4;
#pragma unroll
  for (int q = 0; q < 4; ++q) {
    int n  = (nt0 + q) * 16 + (lane & 15);
    float bv = bias[n];
#pragma unroll
    for (int v = 0; v < 8; ++v) {
      int row = mt * 16 + v + 8 * hi;
      zx[(size_t)row * GATES + n] = acc[q][v] + bv;
    }
  }
}

// --- Phase B: cooperative sequential scan ----------------------------------
__global__ __launch_bounds__(256) void lstm_scan(
    const float* __restrict__ zx, const unsigned short* __restrict__ whf,
    const unsigned char* __restrict__ term, const float* __restrict__ c0,
    unsigned short* __restrict__ hbuf, unsigned* __restrict__ bar,
    float* __restrict__ out) {
  __shared__ __attribute__((aligned(32))) unsigned short whl[8 * 16 * 512]; // 128 KiB Wh slice
  __shared__ __attribute__((aligned(32))) unsigned short hfrag[512];        // h as A-frag image
  __shared__ float zbuf[128];
  __shared__ float cbuf[32];

  const int j    = blockIdx.x;           // WG owns hidden units [32j, 32j+32)
  const int tid  = threadIdx.x;
  const int wave = tid >> 5;
  const int lane = tid & 31;
  const int hi   = lane >> 4;

  // ---- Preload this WG's 8 global N-tiles of packed Wh into LDS (128 KiB).
  // gate columns: {32j+i, 512+32j+i, 1024+32j+i, 1536+32j+i} -> tiles:
  // gt(q) = 2j + (q&1) + (q>>1)*32, q = 0..7.  As a 2D TDM tile this is
  // 4 rows x 32 KiB with a 512 KiB row stride, dest contiguous in LDS.
#if __has_builtin(__builtin_amdgcn_tensor_load_to_lds)
  if (wave == 0) {
    unsigned lds_base = (unsigned)(uintptr_t)&whl[0];
    unsigned long long ga =
        (unsigned long long)(uintptr_t)(whf + (size_t)j * 16384); // +j*32KiB
    // D# group 0: count=1(valid), lds_addr, 57b global_addr, type=2
    v4u g0;
    g0.x = 1u;
    g0.y = lds_base;
    g0.z = (unsigned)(ga & 0xFFFFFFFFu);
    g0.w = (unsigned)((ga >> 32) & 0x01FFFFFFu) | 0x80000000u;
    // D# group 1: data_size=4B(code 2); tensor_dim0=131072, tensor_dim1=128;
    // tile_dim0=8192 dwords (32 KiB), tile_dim1=4; dim0_stride=131072 dwords.
    v8i g1;
    g1[0] = (int)(2u << 16);                          // mask=0 | data_size=2
    g1[1] = (int)((131072u & 0xFFFFu) << 16);         // bar=0 | dim0 lo16 (=0)
    g1[2] = (int)((131072u >> 16) | (128u << 16));    // dim0 hi16 | dim1 lo16
    g1[3] = (int)(8192u << 16);                       // dim1 hi16 | tile_dim0
    g1[4] = 4;                                        // tile_dim1=4, tile_dim2=0
    g1[5] = (int)131072u;                             // dim0_stride lo32
    g1[6] = 0;                                        // stride hi | dim1_stride lo
    g1[7] = 0;                                        // dim1_stride hi
    v4i gz4 = {0, 0, 0, 0};
    v8i gz8 = {0, 0, 0, 0, 0, 0, 0, 0};
    __builtin_amdgcn_tensor_load_to_lds(g0, g1, gz4, gz4, gz8, 0);
    __builtin_amdgcn_s_wait_tensorcnt(0);
  }
#else
#pragma unroll
  for (int q = 0; q < 8; ++q) {
    int gt = 2 * j + (q & 1) + (q >> 1) * 32;
    const unsigned short* src = whf + (size_t)gt * 16 * 512;
#pragma unroll
    for (int i = 0; i < 4; ++i) {
      int off = tid * 8 + i * 2048;
      *(uint4*)&whl[q * 8192 + off] = *(const uint4*)&src[off];
    }
  }
#endif
  if (tid < 32) cbuf[tid] = c0[j * 32 + tid];
  __syncthreads();

  const int mygt = 2 * j + (wave & 1) + (wave >> 1) * 32;  // this wave's N-tile
  const int ncol = mygt * 16 + (lane & 15);

  float* out_hs  = out;
  float* out_cs  = out + (size_t)T_STEPS * DMODEL;
  float* out_hs2 = out + (size_t)2 * T_STEPS * DMODEL;

  for (int t = 0; t < T_STEPS; ++t) {
    const int termv = (int)term[t];

    // Zx splat load first: its VMEM latency overlaps the hfrag build/barrier.
    float zi = zx[(size_t)t * GATES + ncol];

    // Build replicated-row A-fragment image of keep*h_{t-1} in LDS.
    // k -> (kt = k>>5, r = k&31): hi = (r>>3)&1 ; e = (r&7) + 8*(r>>4)
    const unsigned short* hb = hbuf + (t & 1) * DMODEL;
#pragma unroll
    for (int s = 0; s < 2; ++s) {
      int k  = tid + s * 256;
      int kt = k >> 5, r = k & 31;
      int hh = (r >> 3) & 1;
      int e  = (r & 7) + 8 * (r >> 4);
      hfrag[(kt * 2 + hh) * 16 + e] = termv ? (unsigned short)0 : hb[k];
    }
    __syncthreads();

    // z = Zx[t] + h @ Wh : splat Zx into C, two independent 8-deep chains
    v8f acc0 = {zi, zi, zi, zi, zi, zi, zi, zi};
    v8f acc1 = {};
#pragma unroll
    for (int kt = 0; kt < 8; ++kt) {
      v16bf a0 = *(const v16bf*)&hfrag[(kt * 2 + hi) * 16];
      v16bf b0 = *(const v16bf*)&whl[wave * 8192 + kt * 512 + lane * 16];
      acc0 = wmma_bf16(a0, b0, acc0);
      v16bf a1 = *(const v16bf*)&hfrag[((kt + 8) * 2 + hi) * 16];
      v16bf b1 = *(const v16bf*)&whl[wave * 8192 + (kt + 8) * 512 + lane * 16];
      acc1 = wmma_bf16(a1, b1, acc1);
    }
    if (lane < 16)
      zbuf[(wave >> 1) * 32 + (wave & 1) * 16 + lane] = acc0[0] + acc1[0];

    if (t + 8 < T_STEPS)
      __builtin_prefetch(zx + (size_t)(t + 8) * GATES + ncol, 0, 3);
    __syncthreads();

    // Gate math on 32 owned hidden units; stage next h (bf16) through L2.
    if (tid < 32) {
      float gi = zbuf[tid];
      float gf = zbuf[32 + tid];
      float gg = zbuf[64 + tid];
      float go = zbuf[96 + tid];
      float cp = termv ? 0.f : cbuf[tid];
      float si = 1.f / (1.f + __expf(-gi));
      float sf = 1.f / (1.f + __expf(-gf));
      float so = 1.f / (1.f + __expf(-go));
      float tg = tanhf(gg);
      float nc = sf * cp + si * tg;
      float nh = so * tanhf(nc);
      int d = j * 32 + tid;
      size_t o = (size_t)t * DMODEL + d;
      out_hs[o]  = nh;
      out_cs[o]  = nc;
      out_hs2[o] = nh;
      cbuf[tid] = nc;
      hbuf[((t + 1) & 1) * DMODEL + d] = f2bf(nh);
    }
    __threadfence();
    __syncthreads();

    // Device-scope sense-reversing barrier across the 16 workgroups.
    if (tid == 0) {
      unsigned arrived = __hip_atomic_fetch_add(&bar[0], 1u, __ATOMIC_ACQ_REL,
                                                __HIP_MEMORY_SCOPE_AGENT);
      if (arrived == NWG - 1) {
        __hip_atomic_store(&bar[0], 0u, __ATOMIC_RELAXED, __HIP_MEMORY_SCOPE_AGENT);
        __hip_atomic_store(&bar[1], (unsigned)(t + 1), __ATOMIC_RELEASE,
                           __HIP_MEMORY_SCOPE_AGENT);
      } else {
        while (__hip_atomic_load(&bar[1], __ATOMIC_ACQUIRE,
                                 __HIP_MEMORY_SCOPE_AGENT) <= (unsigned)t)
          __builtin_amdgcn_s_sleep(1);
      }
    }
    __syncthreads();
  }
}

extern "C" void kernel_launch(void* const* d_in, const int* in_sizes, int n_in,
                              void* d_out, int out_size, void* d_ws, size_t ws_size,
                              hipStream_t stream) {
  const float*         X    = (const float*)d_in[0];
  const unsigned char* term = (const unsigned char*)d_in[1];
  const float*         c0   = (const float*)d_in[2];
  const float*         h0   = (const float*)d_in[3];
  const float*         Wx   = (const float*)d_in[4];
  const float*         Wh   = (const float*)d_in[5];
  const float*         b    = (const float*)d_in[6];
  float* out = (float*)d_out;

  char* ws = (char*)d_ws;
  float*          zx   = (float*)(ws + OFF_ZX);
  unsigned short* xf   = (unsigned short*)(ws + OFF_XF);
  unsigned short* wxf  = (unsigned short*)(ws + OFF_WXF);
  unsigned short* whf  = (unsigned short*)(ws + OFF_WHF);
  unsigned short* hbuf = (unsigned short*)(ws + OFF_HBUF);
  unsigned*       bar  = (unsigned*)(ws + OFF_BAR);

  lstm_pack_x<<<(T_STEPS * DMODEL) / 256, 256, 0, stream>>>(X, xf);
  lstm_pack_w<<<(DMODEL * GATES) / 256, 256, 0, stream>>>(Wx, wxf);
  lstm_pack_w<<<(DMODEL * GATES) / 256, 256, 0, stream>>>(Wh, whf);
  lstm_init<<<2, 256, 0, stream>>>(h0, hbuf, bar);
  lstm_gemm_xw<<<dim3(T_STEPS / 16, 4), 256, 0, stream>>>(xf, wxf, b, zx);
  lstm_scan<<<NWG, 256, 0, stream>>>(zx, whf, term, c0, hbuf, bar, out);
}
